// MoDLReconOneUnroll_36283883717150
// MI455X (gfx1250) — matrix operand order repaired and checked by reading, not compiled, not run on hardware
//
#include <hip/hip_runtime.h>
#include <hip/hip_bf16.h>
#include <cstdint>
#include <cstddef>

// ---------------------------------------------------------------------------
// MoDL recon (one unroll) for MI455X / gfx1250, wave32.
//  - SENSE ops: single-wave 320-pt FFT (radix-10 per lane x 32-pt cross-lane
//    shuffle FFT), centered/ortho folded in, mask fused into first inverse pass.
//  - Denoiser layer2 (64->64, the 60-GFLOP hotspot): implicit GEMM with
//    v_wmma_f32_16x16x32_f16; B tile staged into LDS by the Tensor Data Mover
//    (tensor_load_to_lds + s_wait_tensorcnt), input kept with a 1-px zero halo
//    so the TDM tile is a pure strided 2D copy.
//  - CG: elementwise + block-reduction kernels, per-batch scalars in d_ws.
// ---------------------------------------------------------------------------

typedef __attribute__((ext_vector_type(16))) _Float16 v16h;
typedef __attribute__((ext_vector_type(8)))  _Float16 v8h;
typedef __attribute__((ext_vector_type(8)))  float    v8f;
typedef __attribute__((ext_vector_type(4)))  uint32_t u32x4;
typedef __attribute__((ext_vector_type(8)))  uint32_t u32x8;

#define L2LAM    0.05f
#define CG_ITERS 6
#define B_   8
#define C_   12
#define H_   320
#define W_   320
#define HW_  (H_*W_)
#define BHW_ (B_*HW_)
#define BCHW_ (B_*C_*HW_)
#define PW_  322                    // padded width/height for h1 halo
#define H1P_ELEMS ((size_t)PW_*PW_*64)

__device__ __forceinline__ float2 cmulf(float2 a, float2 b) {
  return make_float2(a.x*b.x - a.y*b.y, a.x*b.y + a.y*b.x);
}

// cos/sin(2*pi*j/10), j=0..9
__constant__ float C10c[10] = { 1.f,  0.80901699f,  0.30901699f, -0.30901699f, -0.80901699f,
                               -1.f, -0.80901699f, -0.30901699f,  0.30901699f,  0.80901699f};
__constant__ float S10c[10] = { 0.f,  0.58778525f,  0.95105652f,  0.95105652f,  0.58778525f,
                                0.f, -0.58778525f, -0.95105652f, -0.95105652f, -0.58778525f};

// ---------------------------------------------------------------------------
// Centered orthonormal 320-pt FFT along a line; one wave per line, 4 lines per
// 128-thread block. DIR = -1 forward, +1 inverse.  N = 10*32:
//   n = 32*n1 + lane, k = k1 + 10*k2;  per-lane DFT-10, twiddle W320^(lane*k1),
//   cross-lane DIF-32 via shfl_xor + bit-reversal shuffle.
// Centered transform: *(-1)^n on load, *(-1)^k and 1/sqrt(320) on store.
// Optional k-space mask multiply fused into the load.
// ---------------------------------------------------------------------------
template<int DIR>
__global__ void fft320_kernel(const float2* __restrict__ src, float2* __restrict__ dst,
                              const float* __restrict__ mask,
                              int num_lines, int lines_per_image, int line_stride,
                              int image_stride, int elem_stride)
{
  const int lane = threadIdx.x & 31;
  const int line = blockIdx.x * 4 + (threadIdx.x >> 5);
  if (line >= num_lines) return;
  const int img = line / lines_per_image;
  const int lin = line - img * lines_per_image;
  const size_t base = (size_t)img * image_stride + (size_t)lin * line_stride;
  const int b = img / C_;
  const int mb = b * HW_ + lin * line_stride;

  float2 v[10];
  const float isgn = (lane & 1) ? -1.f : 1.f;
#pragma unroll
  for (int n1 = 0; n1 < 10; ++n1) {
    int pos = 32*n1 + lane;
    float2 t = src[base + (size_t)pos * elem_stride];
    if (mask) { float m = mask[mb + pos * elem_stride]; t.x *= m; t.y *= m; }
    v[n1] = make_float2(t.x * isgn, t.y * isgn);
  }

  // per-lane 10-point DFT over n1
  float2 A[10];
#pragma unroll
  for (int k1 = 0; k1 < 10; ++k1) {
    float sx = 0.f, sy = 0.f;
#pragma unroll
    for (int n1 = 0; n1 < 10; ++n1) {
      const int j = (n1 * k1) % 10;
      const float cr = C10c[j];
      const float ci = (float)DIR * S10c[j];
      sx += v[n1].x * cr - v[n1].y * ci;
      sy += v[n1].x * ci + v[n1].y * cr;
    }
    A[k1] = make_float2(sx, sy);
  }

  // twiddle: A[k1] *= exp(DIR*i*2*pi*lane*k1/320), built incrementally
  {
    float ang = (float)DIR * 6.28318530718f * (float)lane / 320.f;
    float sw, cw; __sincosf(ang, &sw, &cw);
    float tx = 1.f, ty = 0.f;
#pragma unroll
    for (int k1 = 0; k1 < 10; ++k1) {
      float ax = A[k1].x, ay = A[k1].y;
      A[k1] = make_float2(ax*tx - ay*ty, ax*ty + ay*tx);
      float ntx = tx*cw - ty*sw;
      ty = tx*sw + ty*cw; tx = ntx;
    }
  }

  // cross-lane 32-pt DIF FFT (5 stages), all 10 rows per stage
#pragma unroll
  for (int h = 16; h >= 1; h >>= 1) {
    const int j = lane & (h - 1);
    float ang = (float)DIR * 6.28318530718f * (float)j / (float)(2*h);
    float ss, cs; __sincosf(ang, &ss, &cs);
    const bool hi = (lane & h) != 0;
#pragma unroll
    for (int r = 0; r < 10; ++r) {
      float ux = __shfl_xor(A[r].x, h);
      float uy = __shfl_xor(A[r].y, h);
      float nx, ny;
      if (!hi) { nx = A[r].x + ux; ny = A[r].y + uy; }
      else {
        float dx = ux - A[r].x, dy = uy - A[r].y;
        nx = dx*cs - dy*ss; ny = dx*ss + dy*cs;
      }
      A[r] = make_float2(nx, ny);
    }
  }
  // undo bit-reversal across lanes
  const int rl = __brev(lane) >> 27;
#pragma unroll
  for (int r = 0; r < 10; ++r) {
    A[r].x = __shfl(A[r].x, rl);
    A[r].y = __shfl(A[r].y, rl);
  }

  const float scale = 0.0559016994f;  // 1/sqrt(320)
#pragma unroll
  for (int k1 = 0; k1 < 10; ++k1) {
    float s2 = (k1 & 1) ? -scale : scale;
    dst[base + (size_t)(10*lane + k1) * elem_stride] = make_float2(A[k1].x*s2, A[k1].y*s2);
  }
}

// ---------------------------------------------------------------------------
// coil[b,c,:] = maps[b,c,:] * x[b,:]
__global__ void coilmul_kernel(const float2* __restrict__ maps, const float2* __restrict__ xin,
                               float2* __restrict__ out)
{
  int i = blockIdx.x * 256 + threadIdx.x;
  if (i >= BCHW_) return;
  int sp = i % HW_;
  int b  = i / (C_ * HW_);
  out[i] = cmulf(maps[i], xin[b * HW_ + sp]);
}

// out[b,:] = sum_c conj(maps[b,c,:]) * coil[b,c,:]  (+ lam * lam_in)
__global__ void combine_kernel(const float2* __restrict__ maps, const float2* __restrict__ coil,
                               const float2* __restrict__ lam_in, float lam,
                               float2* __restrict__ out)
{
  int i = blockIdx.x * 256 + threadIdx.x;
  if (i >= BHW_) return;
  int sp = i % HW_;
  int b  = i / HW_;
  float sx = 0.f, sy = 0.f;
#pragma unroll 4
  for (int c = 0; c < C_; ++c) {
    size_t idx = ((size_t)(b * C_ + c)) * HW_ + sp;
    float2 m = maps[idx], v = coil[idx];
    sx += m.x*v.x + m.y*v.y;   // Re(conj(m)*v)
    sy += m.x*v.y - m.y*v.x;   // Im(conj(m)*v)
  }
  if (lam_in) { float2 l = lam_in[i]; sx += lam*l.x; sy += lam*l.y; }
  out[i] = make_float2(sx, sy);
}

// ---------------------------------------------------------------------------
// zero a dword region (used once per launch for the h1 halo buffer)
__global__ void zero32_kernel(uint32_t* __restrict__ p, int n)
{
  int i = blockIdx.x * 256 + threadIdx.x;
  if (i < n) p[i] = 0u;
}

// Denoiser layer 1: 2 -> 64, ReLU, channel-last f16 into padded (322x322) image
__global__ void conv1_kernel(const float2* __restrict__ xin, const float* __restrict__ w1,
                             const float* __restrict__ b1, _Float16* __restrict__ h1p)
{
  int i = blockIdx.x * 256 + threadIdx.x;
  if (i >= HW_) return;
  int x = i % W_, y = i / W_;
  float nb[3][3][2];
#pragma unroll
  for (int ry = 0; ry < 3; ++ry)
#pragma unroll
    for (int rx = 0; rx < 3; ++rx) {
      int yg = y + ry - 1, xg = x + rx - 1;
      float2 v = make_float2(0.f, 0.f);
      if ((unsigned)yg < H_ && (unsigned)xg < W_) v = xin[yg * W_ + xg];
      nb[ry][rx][0] = v.x; nb[ry][rx][1] = v.y;
    }
  _Float16* outp = &h1p[(((size_t)(y + 1)) * PW_ + (x + 1)) * 64];
  for (int co = 0; co < 64; ++co) {
    float s = b1[co];
#pragma unroll
    for (int ci = 0; ci < 2; ++ci)
#pragma unroll
      for (int ry = 0; ry < 3; ++ry)
#pragma unroll
        for (int rx = 0; rx < 3; ++rx)
          s += w1[((co*2 + ci)*3 + ry)*3 + rx] * nb[ry][rx][ci];
    outp[co] = (_Float16)fmaxf(s, 0.f);
  }
}

// Reorder w2[64][64][3][3] (f32) into WMMA A-fragment order (f16):
// wfrag[co_tile(4)][t(18)][lane(32)][e(16)], K index = ky*192 + kx*64 + ci.
__global__ void wreorder_kernel(const float* __restrict__ w2, _Float16* __restrict__ wfrag)
{
  int i = blockIdx.x * 256 + threadIdx.x;
  if (i >= 4*18*32*16) return;
  int e    = i & 15;
  int lane = (i >> 4) & 31;
  int t    = (i >> 9) % 18;
  int cot  = (i >> 9) / 18;
  int khalf = lane >> 4;
  int kl = e + 16*khalf;
  int K  = t * 32 + kl;
  int ci = K & 63;
  int kyx = K >> 6;
  int ky = kyx / 3, kx = kyx - 3*ky;
  int co = cot * 16 + (lane & 15);
  wfrag[i] = (_Float16)w2[((co*64 + ci)*3 + ky)*3 + kx];
}

// Denoiser layer 2: 64 -> 64 implicit GEMM with v_wmma_f32_16x16x32_f16.
// Block: 256 thr = 8 waves, tile = 64 c_out x 32 pixels (one row segment).
// B tile (3 x 34 x 64 f16 halo slab) fetched by the Tensor Data Mover:
// 2D tile, row chunk 4352 B, row stride 41216 B, 3 rows -> 13056 B in LDS.
__global__ void conv2_wmma_kernel(const _Float16* __restrict__ h1p,
                                  const _Float16* __restrict__ wfrag,
                                  const float* __restrict__ b2,
                                  _Float16* __restrict__ h2)
{
  __shared__ __attribute__((aligned(32))) _Float16 lds[3 * 34 * 64];  // [ry][cx][ci]
  const int x0  = blockIdx.x * 32;
  const int y   = blockIdx.y;
  const int tid = threadIdx.x;

  // --- TDM: async tensor load of the halo tile into LDS (wave 0 issues) ---
  if (tid < 32) {
    const uint32_t lds_off = (uint32_t)(uintptr_t)(&lds[0]);
    const uint64_t ga = (uint64_t)(uintptr_t)&h1p[((size_t)y * PW_ + x0) * 64];
    // D# group 0: count=1 | lds_addr | global_addr | type=2 ("image")
    u32x4 g0 = { 1u,
                 lds_off,
                 (uint32_t)ga,
                 ((uint32_t)(ga >> 32) & 0x01FFFFFFu) | (2u << 30) };
    // D# group 1: data_size=8B; tensor dims huge (no OOB); tile 544x3 elems;
    // dim0 stride = 5152 elems (= 322*64*2 B)
    u32x8 g1 = { (3u << 16),                 // data_size = 3 (8 bytes)
                 0xFFFF0000u,                // tensor_dim0[15:0] << 16
                 0x00007FFFu | 0xFFFF0000u,  // tensor_dim0[31:16] | tensor_dim1[15:0]<<16
                 0x00007FFFu | (544u << 16), // tensor_dim1[31:16] | tile_dim0=544
                 3u,                         // tile_dim1 = 3 (tile_dim2 = 0)
                 5152u,                      // tensor_dim0_stride lo
                 0u, 0u };
    u32x4 gz = { 0u, 0u, 0u, 0u };
    asm volatile("tensor_load_to_lds %0, %1, %2, %3"
                 :: "s"(g0), "s"(g1), "s"(gz), "s"(gz)
                 : "memory");
    __builtin_amdgcn_s_wait_tensorcnt(0);
  }
  __syncthreads();

  const int wave = tid >> 5;
  const int lane = tid & 31;
  const int wm = wave & 3;          // c_out tile (16 rows)
  const int wn = wave >> 2;         // pixel tile (16 cols)
  const int lh = lane >> 4;
  const int pxl = wn * 16 + (lane & 15);

  __builtin_prefetch(&wfrag[((size_t)wm * 18 * 32 + lane) * 16], 0, 1);

  v8f acc = {};
#pragma unroll
  for (int t = 0; t < 18; ++t) {
    const int kyx = t >> 1;
    const int ky = kyx / 3;
    const int kx = kyx - 3 * ky;
    // A fragment: pre-swizzled weights, 32B contiguous per lane
    const v16h a = *(const v16h*)&wfrag[(((size_t)wm * 18 + t) * 32 + lane) * 16];
    // B fragment: lane<16 -> K 0..15 of chunk, lane>=16 -> K 16..31; K=ci here
    const int cib = ((t & 1) << 5) + (lh << 4);
    const v16h bm = *(const v16h*)&lds[(ky * 34 + pxl + kx) * 64 + cib];
    acc = __builtin_amdgcn_wmma_f32_16x16x32_f16(false, a, false, bm,
                                                 (short)0, acc, false, false);
  }

  // D layout: lane<16 -> (M=r, N=lane); lane>=16 -> (M=8+r, N=lane-16)
  const int co0 = wm * 16 + lh * 8;
  size_t obase = ((size_t)y * W_ + (x0 + pxl)) * 64 + co0;
#pragma unroll
  for (int r = 0; r < 8; ++r) {
    float v = acc[r] + b2[co0 + r];
    h2[obase + r] = (_Float16)fmaxf(v, 0.f);
  }
}

// Denoiser layer 3 + residual + rhs:  rhs = x_adj + lam*(xr + conv3(h2))
__global__ void conv3_rhs_kernel(const _Float16* __restrict__ h2, const float* __restrict__ w3,
                                 const float* __restrict__ b3, const float2* __restrict__ xin,
                                 const float2* __restrict__ xadj, float2* __restrict__ rhs)
{
  int i = blockIdx.x * 256 + threadIdx.x;
  if (i >= HW_) return;
  int x = i % W_, y = i / W_;
  float a0 = b3[0], a1 = b3[1];
  for (int ry = 0; ry < 3; ++ry) {
    int yg = y + ry - 1; if ((unsigned)yg >= H_) continue;
    for (int rx = 0; rx < 3; ++rx) {
      int xg = x + rx - 1; if ((unsigned)xg >= W_) continue;
      const _Float16* hp = &h2[((size_t)yg * W_ + xg) * 64];
      const int kk = ry * 3 + rx;
#pragma unroll 16
      for (int ci = 0; ci < 64; ++ci) {
        float hv = (float)hp[ci];
        a0 += w3[ci * 9 + kk] * hv;         // w3[0][ci][ky][kx]
        a1 += w3[576 + ci * 9 + kk] * hv;   // w3[1][ci][ky][kx]
      }
    }
  }
  float2 xr = xin[i];
  float2 xa = xadj[i];
  float r0 = xr.x + a0, r1 = xr.y + a1;
  rhs[i] = make_float2(xa.x + L2LAM * r0, xa.y + L2LAM * r1);
}

// ---------------------------------------------------------------------------
// CG support kernels
__global__ void rinit_kernel(const float2* __restrict__ rhs, const float2* __restrict__ Ap,
                             float2* __restrict__ r, float2* __restrict__ p)
{
  int i = blockIdx.x * 256 + threadIdx.x;
  if (i >= BHW_) return;
  float2 t = make_float2(rhs[i].x - Ap[i].x, rhs[i].y - Ap[i].y);
  r[i] = t; p[i] = t;
}

__global__ void zdot_kernel(const float2* __restrict__ a, const float2* __restrict__ b,
                            float* __restrict__ out)
{
  const int bb = blockIdx.y;
  const float2* pa = a + (size_t)bb * HW_;
  const float2* pb = b + (size_t)bb * HW_;
  float s = 0.f;
  for (int i = blockIdx.x * blockDim.x + threadIdx.x; i < HW_; i += gridDim.x * blockDim.x) {
    float2 va = pa[i], vb = pb[i];
    s += va.x * vb.x + va.y * vb.y;
  }
  __shared__ float red[256];
  red[threadIdx.x] = s;
  __syncthreads();
  for (int o = 128; o > 0; o >>= 1) {
    if (threadIdx.x < o) red[threadIdx.x] += red[threadIdx.x + o];
    __syncthreads();
  }
  if (threadIdx.x == 0) atomicAdd(&out[bb], red[0]);
}

__global__ void zero8_kernel(float* __restrict__ p)
{ if (threadIdx.x < 8) p[threadIdx.x] = 0.f; }

__global__ void cg_alpha_kernel(const float* __restrict__ rs, const float* __restrict__ pap,
                                float* __restrict__ alpha)
{ int i = threadIdx.x; if (i < 8) alpha[i] = rs[i] / pap[i]; }

__global__ void cg_beta_kernel(float* __restrict__ rs, const float* __restrict__ rsn,
                               float* __restrict__ beta)
{ int i = threadIdx.x; if (i < 8) { beta[i] = rsn[i] / rs[i]; rs[i] = rsn[i]; } }

__global__ void update_xr_kernel(float2* __restrict__ x, float2* __restrict__ r,
                                 const float2* __restrict__ p, const float2* __restrict__ Ap,
                                 const float* __restrict__ alpha)
{
  int i = blockIdx.x * 256 + threadIdx.x;
  if (i >= BHW_) return;
  float a = alpha[i / HW_];
  float2 xv = x[i], rv = r[i], pv = p[i], av = Ap[i];
  x[i] = make_float2(xv.x + a * pv.x, xv.y + a * pv.y);
  r[i] = make_float2(rv.x - a * av.x, rv.y - a * av.y);
}

__global__ void update_p_kernel(float2* __restrict__ p, const float2* __restrict__ r,
                                const float* __restrict__ beta)
{
  int i = blockIdx.x * 256 + threadIdx.x;
  if (i >= BHW_) return;
  float bt = beta[i / HW_];
  float2 pv = p[i], rv = r[i];
  p[i] = make_float2(rv.x + bt * pv.x, rv.y + bt * pv.y);
}

// ---------------------------------------------------------------------------
// Host-side helpers
static void fft_pass(hipStream_t s, const float2* src, float2* dst, const float* mask,
                     bool inverse, bool rows)
{
  const int num_lines = B_ * C_ * H_;   // H_ == W_
  dim3 grid((num_lines + 3) / 4);
  const int ls = rows ? W_ : 1;
  const int es = rows ? 1 : W_;
  if (inverse)
    fft320_kernel<+1><<<grid, 128, 0, s>>>(src, dst, mask, num_lines, 320, ls, HW_, es);
  else
    fft320_kernel<-1><<<grid, 128, 0, s>>>(src, dst, mask, num_lines, 320, ls, HW_, es);
}

// Aop(p) = A^H A p + lam*p
static void run_aop(hipStream_t s, const float2* maps, const float* masks,
                    const float2* pin, float2* coil, float2* out)
{
  coilmul_kernel<<<(BCHW_ + 255) / 256, 256, 0, s>>>(maps, pin, coil);
  fft_pass(s, coil, coil, nullptr, false, true);    // fftc rows
  fft_pass(s, coil, coil, nullptr, false, false);   // fftc cols
  fft_pass(s, coil, coil, masks,   true,  false);   // mask * k, ifftc cols
  fft_pass(s, coil, coil, nullptr, true,  true);    // ifftc rows
  combine_kernel<<<(BHW_ + 255) / 256, 256, 0, s>>>(maps, coil, pin, L2LAM, out);
}

extern "C" void kernel_launch(void* const* d_in, const int* in_sizes, int n_in,
                              void* d_out, int out_size, void* d_ws, size_t ws_size,
                              hipStream_t stream)
{
  const float2* x_in  = (const float2*)d_in[0];
  const float2* maps  = (const float2*)d_in[1];
  const float*  masks = (const float*)d_in[2];
  const float2* ksp   = (const float2*)d_in[3];
  const float*  w1    = (const float*)d_in[4];
  const float*  b1    = (const float*)d_in[5];
  const float*  w2    = (const float*)d_in[6];
  const float*  b2    = (const float*)d_in[7];
  const float*  w3    = (const float*)d_in[8];
  const float*  b3    = (const float*)d_in[9];
  float2* x = (float2*)d_out;   // x lives directly in d_out ([B,H,W,2] == float2)

  // workspace carve-out (~138 MB)
  char* ws = (char*)d_ws;
  size_t off = 0;
  auto take = [&](size_t bytes) -> void* {
    void* p = ws + off;
    off += (bytes + 255) & ~(size_t)255;
    return p;
  };
  float2* coil = (float2*)take((size_t)BCHW_ * sizeof(float2));
  float2* xadj = (float2*)take((size_t)BHW_ * sizeof(float2));
  float2* rhs  = (float2*)take((size_t)BHW_ * sizeof(float2));
  float2* rv   = (float2*)take((size_t)BHW_ * sizeof(float2));
  float2* pv   = (float2*)take((size_t)BHW_ * sizeof(float2));
  float2* Ap   = (float2*)take((size_t)BHW_ * sizeof(float2));
  _Float16* h1p = (_Float16*)take(H1P_ELEMS * sizeof(_Float16));  // padded 322x322x64
  _Float16* h2  = (_Float16*)take((size_t)HW_ * 64 * sizeof(_Float16));
  _Float16* wf  = (_Float16*)take(4 * 18 * 32 * 16 * sizeof(_Float16));
  float* scal  = (float*)take(256);
  float *rs = scal, *pap = scal + 8, *rsn = scal + 16, *alp = scal + 24, *bet = scal + 32;

  // ---- SENSE adjoint: x_adj = A^H(ksp) (mask fused into first inverse pass)
  fft_pass(stream, ksp, coil, masks, true, true);
  fft_pass(stream, coil, coil, nullptr, true, false);
  combine_kernel<<<(BHW_ + 255) / 256, 256, 0, stream>>>(maps, coil, nullptr, 0.f, xadj);

  // ---- denoiser + rhs = x_adj + lam * (x + CNN(x)), batched per image
  {
    const int ndw = (int)(H1P_ELEMS * sizeof(_Float16) / 4);
    zero32_kernel<<<(ndw + 255) / 256, 256, 0, stream>>>((uint32_t*)h1p, ndw);
  }
  wreorder_kernel<<<(4 * 18 * 32 * 16) / 256, 256, 0, stream>>>(w2, wf);
  for (int b = 0; b < B_; ++b) {
    conv1_kernel<<<(HW_ + 255) / 256, 256, 0, stream>>>(x_in + (size_t)b * HW_, w1, b1, h1p);
    conv2_wmma_kernel<<<dim3(W_ / 32, H_), 256, 0, stream>>>(h1p, wf, b2, h2);
    conv3_rhs_kernel<<<(HW_ + 255) / 256, 256, 0, stream>>>(h2, w3, b3,
        x_in + (size_t)b * HW_, xadj + (size_t)b * HW_, rhs + (size_t)b * HW_);
  }

  // ---- CG: x0 = x_in
  hipMemcpyAsync(x, x_in, (size_t)BHW_ * sizeof(float2), hipMemcpyDeviceToDevice, stream);
  run_aop(stream, maps, masks, x, coil, Ap);
  rinit_kernel<<<(BHW_ + 255) / 256, 256, 0, stream>>>(rhs, Ap, rv, pv);
  zero8_kernel<<<1, 32, 0, stream>>>(rs);
  zdot_kernel<<<dim3(64, B_), 256, 0, stream>>>(rv, rv, rs);

  for (int it = 0; it < CG_ITERS; ++it) {
    run_aop(stream, maps, masks, pv, coil, Ap);
    zero8_kernel<<<1, 32, 0, stream>>>(pap);
    zdot_kernel<<<dim3(64, B_), 256, 0, stream>>>(pv, Ap, pap);
    cg_alpha_kernel<<<1, 32, 0, stream>>>(rs, pap, alp);
    update_xr_kernel<<<(BHW_ + 255) / 256, 256, 0, stream>>>(x, rv, pv, Ap, alp);
    zero8_kernel<<<1, 32, 0, stream>>>(rsn);
    zdot_kernel<<<dim3(64, B_), 256, 0, stream>>>(rv, rv, rsn);
    cg_beta_kernel<<<1, 32, 0, stream>>>(rs, rsn, bet);
    update_p_kernel<<<(BHW_ + 255) / 256, 256, 0, stream>>>(pv, rv, bet);
  }

  (void)in_sizes; (void)n_in; (void)out_size; (void)ws_size;
}
